// Attention_64433099374599
// MI455X (gfx1250) — matrix operand order repaired and checked by reading, not compiled, not run on hardware
//
#include <hip/hip_runtime.h>

// ---------------------------------------------------------------------------
// Problem constants (match reference): B=2, T=2048, C=2048, H=16, D=128
// ---------------------------------------------------------------------------
#define BB   2
#define TT   2048
#define CC   2048
#define HH   16
#define DD   128
#define BT   (BB * TT)          // 4096
#define OC   (3 * CC)           // 6144

typedef __bf16 v16bf __attribute__((ext_vector_type(16)));
typedef float  v8f   __attribute__((ext_vector_type(8)));
typedef int    v4i   __attribute__((ext_vector_type(4)));

#if __has_builtin(__builtin_amdgcn_global_load_async_to_lds_b128)
#define HAS_ASYNC_LDS 1
typedef __attribute__((address_space(1))) v4i gbl_v4i;   // global ("__device__")
typedef __attribute__((address_space(3))) v4i lds_v4i;   // LDS ("__shared__")
#else
#define HAS_ASYNC_LDS 0
#endif

// ---------------------------------------------------------------------------
// WMMA helper: D = A(16x32 bf16) * B(32x16 bf16) + C(16x16 f32)
// ---------------------------------------------------------------------------
__device__ __forceinline__ v8f wmma_bf16(v16bf a, v16bf b, v8f c) {
    return __builtin_amdgcn_wmma_f32_16x16x32_bf16(
        /*neg_a=*/false, a, /*neg_b=*/false, b,
        /*c_mod=*/(short)0, c, /*reuse_a=*/false, /*reuse_b=*/false);
}

// A-matrix 16x32 bf16 fragment (this code's k order, same as load_frag_b):
//   row = lane&15 ; lanes<16 carry K 0..15, lanes>=16 carry K 16..31 (contiguous)
__device__ __forceinline__ v16bf load_frag_a(const __bf16* p, int rowStride, int lane) {
    union { v16bf v; uint4 q[2]; } u;
    const __bf16* base = p + (size_t)(lane & 15) * rowStride + ((lane >> 4) << 4);
    u.q[0] = *reinterpret_cast<const uint4*>(base);
    u.q[1] = *reinterpret_cast<const uint4*>(base + 8);
    return u.v;
}

// B-matrix 32x16 bf16 fragment: col n = lane&15 ; lanes<16 hold K=0..15,
// lanes>=16 hold K=16..31 (contiguous per lane). Memory is [n][k] row-major.
__device__ __forceinline__ v16bf load_frag_b(const __bf16* p, int rowStride, int lane) {
    union { v16bf v; uint4 q[2]; } u;
    const __bf16* base = p + (size_t)(lane & 15) * rowStride + ((lane >> 4) << 4);
    u.q[0] = *reinterpret_cast<const uint4*>(base);
    u.q[1] = *reinterpret_cast<const uint4*>(base + 8);
    return u.v;
}

// ---------------------------------------------------------------------------
// Async global -> LDS staging (CDNA5 ASYNCcnt path) with sync fallback
// ---------------------------------------------------------------------------
__device__ __forceinline__ void stage32(const __bf16* g, __bf16* l) {
#if HAS_ASYNC_LDS
    __builtin_amdgcn_global_load_async_to_lds_b128(
        (gbl_v4i*)(g),     (lds_v4i*)(l),     0, 0);
    __builtin_amdgcn_global_load_async_to_lds_b128(
        (gbl_v4i*)(g + 8), (lds_v4i*)(l + 8), 0, 0);
#else
    uint4 a0 = *reinterpret_cast<const uint4*>(g);
    uint4 a1 = *reinterpret_cast<const uint4*>(g + 8);
    *reinterpret_cast<uint4*>(l)     = a0;
    *reinterpret_cast<uint4*>(l + 8) = a1;
#endif
}

template <int N>
__device__ __forceinline__ void wait_async() {
#if HAS_ASYNC_LDS
#if __has_builtin(__builtin_amdgcn_s_wait_asynccnt)
    __builtin_amdgcn_s_wait_asynccnt((unsigned short)N);
#else
    asm volatile("s_wait_asynccnt %0" :: "i"(N) : "memory");
#endif
#endif
}

// ---------------------------------------------------------------------------
// f32 -> bf16 conversion
// ---------------------------------------------------------------------------
__global__ void cvt_f32_bf16_kernel(const float* __restrict__ src,
                                    __bf16* __restrict__ dst, int n) {
    int i = blockIdx.x * blockDim.x + threadIdx.x;
    int stride = gridDim.x * blockDim.x;
    for (; i < n; i += stride) dst[i] = (__bf16)src[i];
}

// ---------------------------------------------------------------------------
// GEMM: Out[M][N] = A[M][K] * Bm[N][K]^T   (bf16 inputs, f32 accumulate)
// Block: 256 threads = 8 waves, tile 128x128, K-step 32, double-buffered
// async global->LDS staging. Wave owns a 32x64 subtile (2x4 WMMA frags).
// ---------------------------------------------------------------------------
template <typename OutT>
__global__ __launch_bounds__(256) void gemm_bf16_kernel(
        const __bf16* __restrict__ A, const __bf16* __restrict__ Bm,
        OutT* __restrict__ Out, int M, int N, int K) {
    __shared__ __align__(16) __bf16 As[2][128][40];
    __shared__ __align__(16) __bf16 Bs[2][128][40];

    const int lane = threadIdx.x & 31;
    const int wave = threadIdx.x >> 5;
    const int m0 = blockIdx.y * 128;
    const int n0 = blockIdx.x * 128;
    const int wm = (wave >> 1) * 32;
    const int wn = (wave & 1) * 64;
    const int lrow  = threadIdx.x >> 1;        // 0..127
    const int lhalf = (threadIdx.x & 1) * 16;  // 0 or 16 (bf16 elems)

    const __bf16* agBase = A  + (size_t)(m0 + lrow) * K + lhalf;
    const __bf16* bgBase = Bm + (size_t)(n0 + lrow) * K + lhalf;

    v8f acc[2][4] = {};

    const int nk = K >> 5;
    // prime buffer 0 (4 async b128 per thread)
    stage32(agBase, &As[0][lrow][lhalf]);
    stage32(bgBase, &Bs[0][lrow][lhalf]);

    for (int it = 0; it < nk; ++it) {
        const int cur = it & 1;
        const bool pf = (it + 1) < nk;
        if (pf) {
            const int koff = (it + 1) << 5;
            stage32(agBase + koff, &As[cur ^ 1][lrow][lhalf]);
            stage32(bgBase + koff, &Bs[cur ^ 1][lrow][lhalf]);
            wait_async<4>();   // newest 4 = prefetch; current tile resolved
        } else {
            wait_async<0>();
        }
        __syncthreads();

        v16bf af[2], bfr[4];
        #pragma unroll
        for (int i = 0; i < 2; ++i) af[i]  = load_frag_a(&As[cur][wm + i * 16][0], 40, lane);
        #pragma unroll
        for (int j = 0; j < 4; ++j) bfr[j] = load_frag_b(&Bs[cur][wn + j * 16][0], 40, lane);
        #pragma unroll
        for (int i = 0; i < 2; ++i)
            #pragma unroll
            for (int j = 0; j < 4; ++j)
                acc[i][j] = wmma_bf16(af[i], bfr[j], acc[i][j]);
        __syncthreads();
    }

    // C/D layout: element r -> row r + 8*(lane>=16), col = lane&15
    const int rbias = (lane >> 4) << 3;
    #pragma unroll
    for (int i = 0; i < 2; ++i)
        #pragma unroll
        for (int j = 0; j < 4; ++j)
            #pragma unroll
            for (int r = 0; r < 8; ++r) {
                int m = m0 + wm + i * 16 + r + rbias;
                int n = n0 + wn + j * 16 + (lane & 15);
                Out[(size_t)m * N + n] = (OutT)acc[i][j][r];
            }
}

// ---------------------------------------------------------------------------
// RoPE + head split.  qkv bf16 [BT][6144] ->
//   q,k bf16 [B][H][T][D]  (rotary applied),  vT bf16 [B][H][D][T]
// ---------------------------------------------------------------------------
__global__ void rope_split_kernel(const __bf16* __restrict__ qkv,
                                  const float* __restrict__ cosb,
                                  const float* __restrict__ sinb,
                                  __bf16* __restrict__ q,
                                  __bf16* __restrict__ k,
                                  __bf16* __restrict__ vt) {
    size_t idx = (size_t)blockIdx.x * blockDim.x + threadIdx.x;
    if (idx >= (size_t)BT * OC) return;
    int col = (int)(idx % OC);
    int bt  = (int)(idx / OC);
    int b = bt >> 11, t = bt & (TT - 1);
    int which = col / CC;            // 0=q 1=k 2=v
    int rest  = col - which * CC;
    int h = rest >> 7, d = rest & (DD - 1);
    float val = (float)qkv[idx];
    size_t hidx = ((size_t)(b * HH + h) * TT + t) * DD + d;
    if (which == 2) {
        vt[((size_t)(b * HH + h) * DD + d) * TT + t] = (__bf16)val;
    } else {
        int d2 = (d < 64) ? d + 64 : d - 64;
        float pv = (float)qkv[(size_t)bt * OC + which * CC + h * DD + d2];
        float rh = (d < 64) ? -pv : pv;
        float outv = val * cosb[t * DD + d] + rh * sinb[t * DD + d];
        if (which == 0) q[hidx] = (__bf16)outv;
        else            k[hidx] = (__bf16)outv;
    }
}

// ---------------------------------------------------------------------------
// Flash attention.  grid = (T/64, B*H), block = 128 (4 waves).
// Each wave owns 16 q-rows; streams 32 k-columns per iteration.
//   S = Q*K^T  : 8 wmma  (K read [t][d] row-major -> natural B frags)
//   P*V        : 8 wmma  (V stored [d][t] -> natural B frags)
// Online softmax state per lane covers its 8 C/D rows.
// ---------------------------------------------------------------------------
__global__ __launch_bounds__(128) void flash_attn_kernel(
        const __bf16* __restrict__ Q, const __bf16* __restrict__ Kc,
        const __bf16* __restrict__ Vt, __bf16* __restrict__ Y) {
    __shared__ __align__(16) __bf16 Ps[4][16][40];

    const int lane = threadIdx.x & 31;
    const int wave = threadIdx.x >> 5;
    const int bh = blockIdx.y;
    const int b = bh >> 4, h = bh & 15;
    const int t0 = blockIdx.x * 64 + wave * 16;
    const float sc = 0.088388347648318447f;   // 1/sqrt(128)

    // Q fragments (16 rows x 128 d) resident in registers
    const __bf16* qbase = Q + ((size_t)bh * TT + t0) * DD;
    v16bf aQ[4];
    #pragma unroll
    for (int s = 0; s < 4; ++s) aQ[s] = load_frag_a(qbase + s * 32, DD, lane);

    v8f o[8] = {};
    float mstat[8], lstat[8];
    #pragma unroll
    for (int r = 0; r < 8; ++r) { mstat[r] = -1.0e30f; lstat[r] = 0.0f; }

    for (int kk = 0; kk < TT; kk += 32) {
        v8f s0 = {}, s1 = {};
        const __bf16* kb0 = Kc + ((size_t)bh * TT + kk) * DD;
        const __bf16* kb1 = kb0 + 16 * DD;
        #pragma unroll
        for (int s = 0; s < 4; ++s) {
            v16bf bk0 = load_frag_b(kb0 + s * 32, DD, lane);
            v16bf bk1 = load_frag_b(kb1 + s * 32, DD, lane);
            s0 = wmma_bf16(aQ[s], bk0, s0);
            s1 = wmma_bf16(aQ[s], bk1, s1);
        }

        // online softmax update over the 32 new columns
        #pragma unroll
        for (int r = 0; r < 8; ++r) {
            float a0 = s0[r] * sc, a1 = s1[r] * sc;
            float vmax = fmaxf(a0, a1);
            #pragma unroll
            for (int off = 8; off >= 1; off >>= 1)
                vmax = fmaxf(vmax, __shfl_xor(vmax, off, 32));
            float mnew = fmaxf(mstat[r], vmax);
            float p0 = __expf(a0 - mnew);
            float p1 = __expf(a1 - mnew);
            float rs = p0 + p1;
            #pragma unroll
            for (int off = 8; off >= 1; off >>= 1)
                rs += __shfl_xor(rs, off, 32);
            float alpha = __expf(mstat[r] - mnew);
            mstat[r] = mnew;
            lstat[r] = lstat[r] * alpha + rs;
            #pragma unroll
            for (int j = 0; j < 8; ++j) o[j][r] *= alpha;
            int m = r + ((lane >> 4) << 3);
            Ps[wave][m][lane & 15]        = (__bf16)p0;
            Ps[wave][m][16 + (lane & 15)] = (__bf16)p1;
        }

        // P (16x32) from per-wave LDS; same-wave DS ops stay in order.
        v16bf aP = load_frag_a(&Ps[wave][0][0], 40, lane);
        const __bf16* vb = Vt + (size_t)bh * DD * TT + kk;
        #pragma unroll
        for (int j = 0; j < 8; ++j) {
            v16bf bv = load_frag_b(vb + (size_t)(j * 16) * TT, TT, lane);
            o[j] = wmma_bf16(aP, bv, o[j]);
        }
    }

    // normalize and store y[b][t][h*128 + d] (bf16)
    const int rbias = (lane >> 4) << 3;
    #pragma unroll
    for (int r = 0; r < 8; ++r) {
        int t = t0 + r + rbias;
        float inv = 1.0f / lstat[r];
        #pragma unroll
        for (int j = 0; j < 8; ++j) {
            Y[((size_t)b * TT + t) * CC + h * DD + j * 16 + (lane & 15)] =
                (__bf16)(o[j][r] * inv);
        }
    }
}

// ---------------------------------------------------------------------------
// Host-side orchestration
// ---------------------------------------------------------------------------
extern "C" void kernel_launch(void* const* d_in, const int* in_sizes, int n_in,
                              void* d_out, int out_size, void* d_ws, size_t ws_size,
                              hipStream_t stream) {
    (void)in_sizes; (void)n_in; (void)out_size; (void)ws_size;
    const float* x     = (const float*)d_in[0];   // [B][T][C]
    const float* cosb  = (const float*)d_in[1];   // [T][D]
    const float* sinb  = (const float*)d_in[2];   // [T][D]
    const float* Wqkv  = (const float*)d_in[3];   // [3C][C]
    const float* Wproj = (const float*)d_in[4];   // [C][C]
    float* out = (float*)d_out;                   // [B][T][C]

    char* ws = (char*)d_ws;
    size_t off = 0;
    __bf16* xw     = (__bf16*)(ws + off); off += (size_t)BT * CC * 2;   // 16.8 MB
    __bf16* wqkvw  = (__bf16*)(ws + off); off += (size_t)OC * CC * 2;   // 25.2 MB
    __bf16* wprojw = (__bf16*)(ws + off); off += (size_t)CC * CC * 2;   //  8.4 MB
    __bf16* qkvw   = (__bf16*)(ws + off); off += (size_t)BT * OC * 2;   // 50.3 MB
    __bf16* qw     = (__bf16*)(ws + off); off += (size_t)BT * CC * 2;   // 16.8 MB
    __bf16* kw     = (__bf16*)(ws + off); off += (size_t)BT * CC * 2;   // 16.8 MB
    __bf16* vtw    = (__bf16*)(ws + off); off += (size_t)BT * CC * 2;   // 16.8 MB
    __bf16* yw     = (__bf16*)(ws + off); off += (size_t)BT * CC * 2;   // 16.8 MB

    // 1) downcast activations + weights to bf16
    cvt_f32_bf16_kernel<<<4096, 256, 0, stream>>>(x,     xw,     BT * CC);
    cvt_f32_bf16_kernel<<<4096, 256, 0, stream>>>(Wqkv,  wqkvw,  OC * CC);
    cvt_f32_bf16_kernel<<<4096, 256, 0, stream>>>(Wproj, wprojw, CC * CC);

    // 2) qkv = x @ Wqkv^T   (M=4096, N=6144, K=2048), bf16 out
    gemm_bf16_kernel<__bf16><<<dim3(OC / 128, BT / 128), 256, 0, stream>>>(
        xw, wqkvw, qkvw, BT, OC, CC);

    // 3) RoPE + split into q,k [B,H,T,D] and vT [B,H,D,T]
    {
        size_t n = (size_t)BT * OC;
        int blocks = (int)((n + 255) / 256);
        rope_split_kernel<<<blocks, 256, 0, stream>>>(qkvw, cosb, sinb, qw, kw, vtw);
    }

    // 4) flash attention -> y bf16 [B,T,C]
    flash_attn_kernel<<<dim3(TT / 64, BB * HH), 128, 0, stream>>>(qw, kw, vtw, yw);

    // 5) out = y @ Wproj^T   (M=4096, N=2048, K=2048), f32 out
    gemm_bf16_kernel<float><<<dim3(CC / 128, BT / 128), 256, 0, stream>>>(
        yw, wprojw, out, BT, CC, CC);
}
